// NeuralConcatenativeSpeechSynthesis_50242527428851
// MI455X (gfx1250) — matrix operand order repaired and checked by reading, not compiled, not run on hardware
//
#include <hip/hip_runtime.h>
#include <hip/hip_bf16.h>

// ---- problem dims (match reference) ----
#define T_DEC 128
#define S_ENC 256
#define BATCH 8
#define EMB   512
#define DEC_H 1024
#define ENC_H 512
#define FEAT  256
#define G3    (3 * DEC_H)   // 3072

typedef __attribute__((ext_vector_type(16))) __bf16 v16bf;
typedef __attribute__((ext_vector_type(8)))  __bf16 v8bf;
typedef __attribute__((ext_vector_type(8)))  float  v8f;
typedef __attribute__((ext_vector_type(4)))  unsigned int v4u;
typedef __attribute__((ext_vector_type(8)))  int    v8i;
typedef __attribute__((ext_vector_type(4)))  int    v4i;

#if __has_builtin(__builtin_amdgcn_tensor_load_to_lds)
#define HAVE_TDM 1
#else
#define HAVE_TDM 0
#endif

// GRU LDS layout (dynamic, bytes): h_bf16 16xD | h_f32 8xD | hg 8x3D | xg_t 8x3D
#define LDS_HBF_OFF 0
#define LDS_HF_OFF  (16 * DEC_H * 2)                       //  32768
#define LDS_HG_OFF  (LDS_HF_OFF + 8 * DEC_H * 4)           //  65536
#define LDS_XG_OFF  (LDS_HG_OFF + 8 * G3 * 4)              // 163840
#define LDS_TOTAL   (LDS_XG_OFF + 8 * G3 * 4)              // 262144

// A-fragment (16x32 bf16), contiguous K: lanes 0-15 K=0..15, hi +16
__device__ __forceinline__ v16bf load_a_frag(const __bf16* __restrict__ p) {
  v8bf lo = *(const v8bf*)p;
  v8bf hi = *(const v8bf*)(p + 8);
  v16bf a;
#pragma unroll
  for (int i = 0; i < 8; ++i) { a[i] = lo[i]; a[8 + i] = hi[i]; }
  return a;
}

// ---------------------------------------------------------------------------
// Vectorized fp32 -> bf16 conversion, 8 elements/thread (n % 2048 == 0).
// ---------------------------------------------------------------------------
__global__ void k_f32_to_bf16(const float* __restrict__ src,
                              __bf16* __restrict__ dst) {
  const int i = (blockIdx.x * blockDim.x + threadIdx.x) * 8;
  const float4 s0 = *(const float4*)(src + i);
  const float4 s1 = *(const float4*)(src + i + 4);
  v8bf d;
  d[0] = (__bf16)s0.x; d[1] = (__bf16)s0.y; d[2] = (__bf16)s0.z; d[3] = (__bf16)s0.w;
  d[4] = (__bf16)s1.x; d[5] = (__bf16)s1.y; d[6] = (__bf16)s1.z; d[7] = (__bf16)s1.w;
  *(v8bf*)(dst + i) = d;
}

// ---------------------------------------------------------------------------
// C[M,N] = A[M,K] * W[N,K]^T, bf16 operands, fp32 accum.
// Register-blocked: each wave owns a 32x64 macro-tile = 2x4 WMMA tiles.
// Per k-step: 2 A-frag + 4 B-frag loads -> 8 v_wmma_f32_16x16x32_bf16.
// ---------------------------------------------------------------------------
__global__ void k_gemm_nt_bf16wmma(const __bf16* __restrict__ A,
                                   const __bf16* __restrict__ W,
                                   float* __restrict__ C,
                                   int M, int N, int K) {
  const int lane = threadIdx.x & 31;
  const int wave = threadIdx.x >> 5;
  const int tile = blockIdx.x * (blockDim.x >> 5) + wave;
  const int nBlkN = N >> 6;                 // 64-wide N blocks
  const int mb = tile / nBlkN;
  const int nb = tile % nBlkN;
  const int l16 = lane & 15;
  const int hi  = lane >> 4;

  int arow[2], bcol[4];
#pragma unroll
  for (int i = 0; i < 2; ++i) arow[i] = (mb << 5) + (i << 4) + l16;
#pragma unroll
  for (int j = 0; j < 4; ++j) bcol[j] = (nb << 6) + (j << 4) + l16;

  v8f acc[2][4];
#pragma unroll
  for (int i = 0; i < 2; ++i)
#pragma unroll
    for (int j = 0; j < 4; ++j) acc[i][j] = (v8f){};

  for (int k0 = 0; k0 < K; k0 += 32) {
    v16bf a[2], b[4];
#pragma unroll
    for (int i = 0; i < 2; ++i)
      a[i] = load_a_frag(A + (size_t)arow[i] * K + k0 + (hi ? 16 : 0));
#pragma unroll
    for (int j = 0; j < 4; ++j) {
      const __bf16* bp = W + (size_t)bcol[j] * K + k0 + (hi ? 16 : 0);
      __builtin_prefetch(bp + 32, 0, 0);     // global_prefetch_b8 next k-slab
      b[j] = *(const v16bf*)bp;
    }
#pragma unroll
    for (int i = 0; i < 2; ++i)
#pragma unroll
      for (int j = 0; j < 4; ++j)
        acc[i][j] = __builtin_amdgcn_wmma_f32_16x16x32_bf16(
            false, a[i], false, b[j], (short)0, acc[i][j], false, false);
  }
#pragma unroll
  for (int i = 0; i < 2; ++i)
#pragma unroll
    for (int j = 0; j < 4; ++j)
#pragma unroll
      for (int r = 0; r < 8; ++r) {
        const int m = (mb << 5) + (i << 4) + r + (hi << 3);
        C[(size_t)m * N + bcol[j]] = acc[i][j][r];
      }
}

// ---------------------------------------------------------------------------
// Persistent single-workgroup GRU scan. 1024 threads = 32 wave32 waves.
// Phase A: hg = h @ W_hh^T (k-outer, A-frag reused across 6 N-tiles/wave).
// xg[t] slab (96KB) TDM-prefetched into LDS before Phase A, consumed in B.
// ---------------------------------------------------------------------------
__global__ void k_gru(const float* __restrict__ xg,       // (T,B,3D) fp32 global
                      const __bf16* __restrict__ Whh,     // (3D,D) bf16
                      float* __restrict__ hs_out,         // (T,B,D) fp32
                      __bf16* __restrict__ hs_bf) {       // (T,B,D) bf16 copy
  extern __shared__ char smem[];
  __bf16* h_bf = (__bf16*)(smem + LDS_HBF_OFF);
  float*  h_f  = (float*)(smem + LDS_HF_OFF);
  float*  hg   = (float*)(smem + LDS_HG_OFF);
  float*  xg_l = (float*)(smem + LDS_XG_OFF);

  const int tid  = threadIdx.x;
  const int lane = tid & 31;
  const int wave = tid >> 5;
  const int l16  = lane & 15;
  const int hi   = lane >> 4;

  for (int i = tid; i < 16 * DEC_H; i += 1024) h_bf[i] = (__bf16)0.0f;
  for (int i = tid; i < 8 * DEC_H;  i += 1024) h_f[i]  = 0.0f;
  __syncthreads();

  int col[6];
#pragma unroll
  for (int nti = 0; nti < 6; ++nti) col[nti] = ((wave + nti * 32) << 4) + l16;

  for (int t = 0; t < T_DEC; ++t) {
    // ---- TDM prefetch of xg[t] (overlaps Phase A WMMA) ----
#if HAVE_TDM
    if (wave == 0) {
      const unsigned long long ga =
          (unsigned long long)(const void*)(xg + (size_t)t * BATCH * G3);
      const unsigned int nelem = BATCH * G3;            // 24576 fp32
      v4u g0; v8i g1; v4i g2 = (v4i){}; v4i g3 = (v4i){};
      g0[0] = 1u;                                       // count=1
      g0[1] = (unsigned int)LDS_XG_OFF;                 // lds_addr (bytes)
      g0[2] = (unsigned int)(ga & 0xffffffffull);       // global_addr[31:0]
      g0[3] = (unsigned int)((ga >> 32) & 0x1ffffffull) // global_addr[56:32]
            | (2u << 30);                               // type=2 (image)
      g1 = (v8i){};
      g1[0] = (int)(2u << 16);                          // data_size=4B
      g1[1] = (int)((nelem & 0xffffu) << 16);           // tensor_dim0[15:0]
      g1[2] = (int)((nelem >> 16) & 0xffffu)            // tensor_dim0[31:16]
            | (int)(1u << 16);                          // tensor_dim1=1
      g1[3] = (int)((nelem & 0xffffu) << 16);           // tile_dim0
      g1[4] = 1;                                        // tile_dim1=1
      g1[5] = (int)nelem;                               // tensor_dim0_stride
#if __clang_major__ >= 23
      v8i g4 = (v8i){};
      __builtin_amdgcn_tensor_load_to_lds(g0, g1, g2, g3, g4, 0);
#else
      __builtin_amdgcn_tensor_load_to_lds(g0, g1, g2, g3, 0);
#endif
    }
#else
    for (int i = tid; i < BATCH * G3; i += 1024)
      xg_l[i] = xg[(size_t)t * BATCH * G3 + i];
#endif

    // ---- Phase A: hg = h @ W_hh^T (WMMA, k-outer) ----
    v8f acc[6];
#pragma unroll
    for (int nti = 0; nti < 6; ++nti) acc[nti] = (v8f){};
    for (int k0 = 0; k0 < DEC_H; k0 += 32) {
      const v16bf a = load_a_frag(h_bf + l16 * DEC_H + k0 + (hi ? 16 : 0));
#pragma unroll
      for (int nti = 0; nti < 6; ++nti) {
        const v16bf b =
            *(const v16bf*)(Whh + (size_t)col[nti] * DEC_H + k0 + (hi ? 16 : 0));
        acc[nti] = __builtin_amdgcn_wmma_f32_16x16x32_bf16(
            false, a, false, b, (short)0, acc[nti], false, false);
      }
    }
#pragma unroll
    for (int nti = 0; nti < 6; ++nti)
#pragma unroll
      for (int r = 0; r < 8; ++r) {
        const int m = r + (hi << 3);
        if (m < BATCH) hg[m * G3 + col[nti]] = acc[nti][r];  // rows 8-15 pad
      }

#if HAVE_TDM
    if (wave == 0) __builtin_amdgcn_s_wait_tensorcnt(0);
#endif
    __syncthreads();

    // ---- Phase B: fused gates from LDS, 8 elements/thread ----
#pragma unroll
    for (int q = 0; q < 8; ++q) {
      const int idx = tid + (q << 10);
      const int bb  = idx >> 10;
      const int j   = idx & (DEC_H - 1);
      const float xr = xg_l[bb * G3 + j];
      const float xz = xg_l[bb * G3 + DEC_H + j];
      const float xn = xg_l[bb * G3 + 2 * DEC_H + j];
      const float hr = hg[bb * G3 + j];
      const float hz = hg[bb * G3 + DEC_H + j];
      const float hn = hg[bb * G3 + 2 * DEC_H + j];
      const float r = 1.0f / (1.0f + __expf(-(xr + hr)));
      const float z = 1.0f / (1.0f + __expf(-(xz + hz)));
      const float n = tanhf(xn + r * hn);
      const float hprev = h_f[bb * DEC_H + j];
      const float hnew  = (1.0f - z) * n + z * hprev;
      h_f[bb * DEC_H + j]  = hnew;
      h_bf[bb * DEC_H + j] = (__bf16)hnew;
      const size_t o = ((size_t)t * BATCH + bb) * DEC_H + j;
      hs_out[o] = hnew;
      hs_bf[o]  = (__bf16)hnew;
    }
    __syncthreads();   // h_bf/h_f RAW, hg & xg_l WAR for next step
  }
}

// ---------------------------------------------------------------------------
// Fused softmax + context (unchanged): one block per (t,b), never
// materializes (T,S,B,E); LDS ds_add_f32 accumulates We[s] = sum_e w.
// ---------------------------------------------------------------------------
__global__ void k_attn_ctx(const float* __restrict__ dproj,   // (T*B, ENC_H)
                           const float* __restrict__ enc_h,   // (S,B,ENC_H)
                           const float* __restrict__ enc_in,  // (S,B,FEAT)
                           float* __restrict__ ctx) {         // (T*B, FEAT)
  __shared__ float We[S_ENC];
  const int tb  = blockIdx.x;
  const int b   = tb & (BATCH - 1);
  const int tid = threadIdx.x;
  We[tid] = 0.0f;
  __syncthreads();

  const float* dp = dproj + (size_t)tb * ENC_H;
#pragma unroll
  for (int eo = 0; eo < ENC_H / 256; ++eo) {
    const int e  = eo * 256 + tid;
    const float de = dp[e];
    float m = -3.4e38f, sum = 0.0f;
    for (int s = 0; s < S_ENC; ++s) {
      const float v = enc_h[(size_t)s * BATCH * ENC_H + b * ENC_H + e] * de;
      const float mo = m;
      m = fmaxf(m, v);
      sum = sum * __expf(mo - m) + __expf(v - m);
    }
    const float inv = 1.0f / sum;
    for (int s = 0; s < S_ENC; ++s) {
      const float v = enc_h[(size_t)s * BATCH * ENC_H + b * ENC_H + e] * de;
      atomicAdd(&We[s], __expf(v - m) * inv);
    }
  }
  __syncthreads();

  float acc = 0.0f;
  for (int s = 0; s < S_ENC; ++s)
    acc += We[s] * enc_in[(size_t)s * BATCH * FEAT + b * FEAT + tid];
  ctx[(size_t)tb * FEAT + tid] = acc;
}

// ---------------------------------------------------------------------------
extern "C" void kernel_launch(void* const* d_in, const int* in_sizes, int n_in,
                              void* d_out, int out_size, void* d_ws, size_t ws_size,
                              hipStream_t stream) {
  const float* dec_in = (const float*)d_in[0];   // (T,B,EMB)
  const float* enc_h  = (const float*)d_in[1];   // (S,B,ENC_H)
  const float* enc_in = (const float*)d_in[2];   // (S,B,FEAT)
  const float* W_ih   = (const float*)d_in[3];   // (3D,EMB)
  const float* W_hh   = (const float*)d_in[4];   // (3D,D)
  const float* W_att  = (const float*)d_in[5];   // (ENC_H,D)

  float* out = (float*)d_out;
  float* hs  = out;
  float* ctx = out + (size_t)T_DEC * BATCH * DEC_H;

  // workspace layout (~27 MB)
  char* ws = (char*)d_ws;
  size_t o = 0;
  float*  xg       = (float*)(ws + o);  o += (size_t)T_DEC * BATCH * G3 * 4; // 12MB
  __bf16* whh_bf   = (__bf16*)(ws + o); o += (size_t)G3 * DEC_H * 2;         //  6MB
  float*  dproj    = (float*)(ws + o);  o += (size_t)T_DEC * BATCH * ENC_H * 4; // 2MB
  __bf16* wih_bf   = (__bf16*)(ws + o); o += (size_t)G3 * EMB * 2;           //  3MB
  __bf16* watt_bf  = (__bf16*)(ws + o); o += (size_t)ENC_H * DEC_H * 2;      //  1MB
  __bf16* decin_bf = (__bf16*)(ws + o); o += (size_t)T_DEC * BATCH * EMB * 2;//  1MB
  __bf16* hs_bf    = (__bf16*)(ws + o);                                      //  2MB

  // 0) one-shot bf16 staging (2048 elems / block)
  k_f32_to_bf16<<<(T_DEC * BATCH * EMB) / 2048, 256, 0, stream>>>(dec_in, decin_bf);
  k_f32_to_bf16<<<(G3 * EMB) / 2048,           256, 0, stream>>>(W_ih,  wih_bf);
  k_f32_to_bf16<<<(G3 * DEC_H) / 2048,         256, 0, stream>>>(W_hh,  whh_bf);
  k_f32_to_bf16<<<(ENC_H * DEC_H) / 2048,      256, 0, stream>>>(W_att, watt_bf);

  // 1) xg = dec_in @ W_ih^T : M=1024 N=3072 K=512; 32x64 macro-tiles /8 waves
  k_gemm_nt_bf16wmma<<<(T_DEC * BATCH / 32) * (G3 / 64) / 8, 256, 0, stream>>>(
      decin_bf, wih_bf, xg, T_DEC * BATCH, G3, EMB);

  // 2) sequential GRU scan (one persistent workgroup, 256KB dynamic LDS)
  k_gru<<<1, 1024, LDS_TOTAL, stream>>>(xg, whh_bf, hs, hs_bf);

  // 3) dproj = hs @ W_att^T : M=1024 N=512 K=1024
  k_gemm_nt_bf16wmma<<<(T_DEC * BATCH / 32) * (ENC_H / 64) / 8, 256, 0, stream>>>(
      hs_bf, watt_bf, dproj, T_DEC * BATCH, ENC_H, DEC_H);

  // 4) fused softmax + context
  k_attn_ctx<<<T_DEC * BATCH, 256, 0, stream>>>(dproj, enc_h, enc_in, ctx);
}
